// SparseFrequencySelector_197568495623
// MI455X (gfx1250) — matrix operand order repaired and checked by reading, not compiled, not run on hardware
//
#include <hip/hip_runtime.h>

typedef __attribute__((ext_vector_type(16))) __bf16 v16bf;
typedef __attribute__((ext_vector_type(8)))  float  v8f;

constexpr int Bn = 128, Nn = 512, Dd = 512, Kk = 8, Hh = 256;
constexpr size_t SEL_OFF  = 0;
constexpr size_t MASK_OFF = (size_t)Bn * Kk * Dd;            // 524288
constexpr size_t IMP_OFF  = MASK_OFF + (size_t)Bn * Nn;      // 589824
constexpr size_t IDX_OFF  = IMP_OFF + (size_t)Bn * Nn;       // 655360

// ---------------------------------------------------------------------------
// Kernel 0: pack W1 (D x H f32 row-major) into bf16 WMMA B-fragment order.
// Fragment index: ((ht*16 + c)*32 + lane)*16 + j
//   lane 0-15  : col = ht*16 + lane,      k = c*32 + j        (K 0..15)
//   lane 16-31 : col = ht*16 + (lane-16), k = c*32 + 16 + j   (K 16..31)
// ---------------------------------------------------------------------------
__global__ void sfs_pack_w1(const float* __restrict__ W1, __bf16* __restrict__ w1f) {
  int id   = blockIdx.x * 256 + threadIdx.x;   // 0 .. 131071
  int j    =  id        & 15;
  int lane = (id >> 4)  & 31;
  int c    = (id >> 9)  & 15;
  int ht   =  id >> 13;
  int col  = ht * 16 + (lane & 15);
  int k    = c * 32 + (lane >> 4) * 16 + j;
  w1f[id] = (__bf16)W1[k * Hh + col];
}

// ---------------------------------------------------------------------------
// Kernel 1: fused  logits = GELU(coeff @ W1 + b1) @ W2 + b2 + base_logits
// 128 threads (4 waves) per block; each wave owns TWO 16-row A tiles (32 rows)
// so every B fragment loaded feeds 2 WMMAs (halves L2 B-traffic vs 1 tile).
// A staged once in LDS (bf16, WMMA A-fragment order); H swept in 4 groups of
// 4 column tiles so only 8 accumulators are live (no spills).
// ---------------------------------------------------------------------------
__global__ void sfs_scorer(const float* __restrict__ coeff,
                           const __bf16* __restrict__ w1f_raw,
                           const float* __restrict__ b1,
                           const float* __restrict__ W2,
                           const float* __restrict__ b2p,
                           const float* __restrict__ base_logits,
                           float* __restrict__ logits) {
  extern __shared__ char smem[];                 // 32 KiB per wave (2 A tiles)
  const int wave  = threadIdx.x >> 5;
  const int lane  = threadIdx.x & 31;
  const int r     = lane & 15;                   // local M row this lane feeds
  const int hi    = lane >> 4;                   // half-wave (K split for A)
  const int col16 = lane & 15;                   // N column within C tile

  const size_t rowBase = (size_t)blockIdx.x * 128 + (size_t)wave * 32;
  char* aBase0 = smem + wave * 32768;            // A tile 0 (rows 0..15)
  char* aBase1 = aBase0 + 16384;                 // A tile 1 (rows 16..31)

  // ---- Stage A: 2 x (16 rows x 512 cols) f32 -> bf16 fragments in LDS -----
  #pragma unroll 1
  for (int mt = 0; mt < 2; ++mt) {
    const float* crow = coeff + (rowBase + mt * 16 + r) * Dd;
    char* aB = mt ? aBase1 : aBase0;
    #pragma unroll 4
    for (int c = 0; c < 16; ++c) {
      const float4* p0 = (const float4*)(crow + c * 32 + hi * 16);
      const float4* p1 = (const float4*)(crow + c * 32 + hi * 16 + 8);
      float4 f0 = p0[0], f1 = p0[1];
      float4 f2 = p1[0], f3 = p1[1];
      v16bf a;
      a[0]  = (__bf16)f0.x;  a[1]  = (__bf16)f0.y;  a[2]  = (__bf16)f0.z;  a[3]  = (__bf16)f0.w;
      a[4]  = (__bf16)f1.x;  a[5]  = (__bf16)f1.y;  a[6]  = (__bf16)f1.z;  a[7]  = (__bf16)f1.w;
      a[8]  = (__bf16)f2.x;  a[9]  = (__bf16)f2.y;  a[10] = (__bf16)f2.z;  a[11] = (__bf16)f2.w;
      a[12] = (__bf16)f3.x;  a[13] = (__bf16)f3.y;  a[14] = (__bf16)f3.z;  a[15] = (__bf16)f3.w;
      *(v16bf*)(aB + c * 1024 + lane * 32) = a;
    }
  }

  const v16bf* w1f = (const v16bf*)w1f_raw;
  v8f part0 = {}, part1 = {};

  // fused epilogue for one finished 16x16 C tile: bias + exact GELU + dot(W2)
  auto epilogue = [&](const v8f& acc, int ht, v8f& part) {
    const float w2v = W2[ht * 16 + col16];
    const float b1v = b1[ht * 16 + col16];
    #pragma unroll
    for (int i = 0; i < 8; ++i) {
      float x = acc[i] + b1v;
      float h = 0.5f * x * (1.0f + erff(x * 0.70710678118654752f));
      part[i] += h * w2v;
    }
  };

  // ---- Sweep H = 16 column tiles in 4 groups of 4 (bounded register use) --
  #pragma unroll 1
  for (int htg = 0; htg < 4; ++htg) {
    asm volatile("" ::: "memory");   // force per-group LDS re-read of A (no LICM)
    v8f acc00 = {}, acc01 = {}, acc02 = {}, acc03 = {};   // row tile 0
    v8f acc10 = {}, acc11 = {}, acc12 = {}, acc13 = {};   // row tile 1
    const v16bf* wbase = w1f + (size_t)(htg * 64) * 32 + lane;
    #pragma unroll 1
    for (int c = 0; c < 16; ++c) {
      v16bf A0 = *(const v16bf*)(aBase0 + c * 1024 + lane * 32);
      v16bf A1 = *(const v16bf*)(aBase1 + c * 1024 + lane * 32);
      v16bf B0 = wbase[(size_t)(0 * 16 + c) * 32];
      v16bf B1 = wbase[(size_t)(1 * 16 + c) * 32];
      v16bf B2 = wbase[(size_t)(2 * 16 + c) * 32];
      v16bf B3 = wbase[(size_t)(3 * 16 + c) * 32];
      acc00 = __builtin_amdgcn_wmma_f32_16x16x32_bf16(false, A0, false, B0, (short)0, acc00, false, false);
      acc10 = __builtin_amdgcn_wmma_f32_16x16x32_bf16(false, A1, false, B0, (short)0, acc10, false, false);
      acc01 = __builtin_amdgcn_wmma_f32_16x16x32_bf16(false, A0, false, B1, (short)0, acc01, false, false);
      acc11 = __builtin_amdgcn_wmma_f32_16x16x32_bf16(false, A1, false, B1, (short)0, acc11, false, false);
      acc02 = __builtin_amdgcn_wmma_f32_16x16x32_bf16(false, A0, false, B2, (short)0, acc02, false, false);
      acc12 = __builtin_amdgcn_wmma_f32_16x16x32_bf16(false, A1, false, B2, (short)0, acc12, false, false);
      acc03 = __builtin_amdgcn_wmma_f32_16x16x32_bf16(false, A0, false, B3, (short)0, acc03, false, false);
      acc13 = __builtin_amdgcn_wmma_f32_16x16x32_bf16(false, A1, false, B3, (short)0, acc13, false, false);
    }
    epilogue(acc00, htg * 4 + 0, part0);  epilogue(acc10, htg * 4 + 0, part1);
    epilogue(acc01, htg * 4 + 1, part0);  epilogue(acc11, htg * 4 + 1, part1);
    epilogue(acc02, htg * 4 + 2, part0);  epilogue(acc12, htg * 4 + 2, part1);
    epilogue(acc03, htg * 4 + 3, part0);  epilogue(acc13, htg * 4 + 3, part1);
  }

  // ---- Reduce partial logits across the 16 lanes (N columns) of each half --
  #pragma unroll
  for (int i = 0; i < 8; ++i) {
    float p = part0[i];
    p += __shfl_xor(p, 1, 16);
    p += __shfl_xor(p, 2, 16);
    p += __shfl_xor(p, 4, 16);
    p += __shfl_xor(p, 8, 16);
    part0[i] = p;
    float q = part1[i];
    q += __shfl_xor(q, 1, 16);
    q += __shfl_xor(q, 2, 16);
    q += __shfl_xor(q, 4, 16);
    q += __shfl_xor(q, 8, 16);
    part1[i] = q;
  }

  if (col16 == 0) {                 // lanes 0 (rows 0-7) and 16 (rows 8-15)
    const float b2 = *b2p;
    #pragma unroll
    for (int i = 0; i < 8; ++i) {
      size_t rowg = rowBase + (size_t)hi * 8 + i;
      int n = (int)(rowg & (Nn - 1));
      logits[rowg] = part0[i] + b2 + base_logits[n];
      size_t rowg1 = rowg + 16;
      int n1 = (int)(rowg1 & (Nn - 1));
      logits[rowg1] = part1[i] + b2 + base_logits[n1];
    }
  }
}

// ---------------------------------------------------------------------------
// Kernel 2: per-batch softmax, iterative top-8 argmax, mask/importance/
// indices/selected-gather. One 256-thread block per batch row (N = 512).
// Logits row is staged into LDS with gfx1250 async global->LDS loads.
// ---------------------------------------------------------------------------
__global__ void sfs_softmax_topk(const float* __restrict__ coeff,
                                 const float* __restrict__ logits,
                                 float* __restrict__ out) {
  __shared__ float sl[512];
  __shared__ float sv[256];
  __shared__ int   si[256];
  __shared__ int   sidx[8];
  const int b = blockIdx.x;
  const int t = threadIdx.x;
  const float* lrow = logits + (size_t)b * Nn;

  // ---- async copy of the 2 KiB logits row into LDS (ASYNCcnt path) -------
  {
    unsigned int       ldsOff = (unsigned int)(uintptr_t)(&sl[2 * t]);
    unsigned long long gaddr  = (unsigned long long)(uintptr_t)(lrow + 2 * t);
    asm volatile("global_load_async_to_lds_b64 %0, %1, off"
                 :: "v"(ldsOff), "v"(gaddr) : "memory");
    asm volatile("s_wait_asynccnt 0" ::: "memory");
  }
  __syncthreads();
  float x0 = sl[t], x1 = sl[t + 256];
  __syncthreads();

  // max-reduce
  sv[t] = fmaxf(x0, x1); __syncthreads();
  for (int s = 128; s > 0; s >>= 1) { if (t < s) sv[t] = fmaxf(sv[t], sv[t + s]); __syncthreads(); }
  float m = sv[0]; __syncthreads();

  // exp + sum-reduce
  float e0 = __expf(x0 - m), e1 = __expf(x1 - m);
  sv[t] = e0 + e1; __syncthreads();
  for (int s = 128; s > 0; s >>= 1) { if (t < s) sv[t] += sv[t + s]; __syncthreads(); }
  float inv = 1.0f / sv[0]; __syncthreads();

  float p0 = e0 * inv, p1 = e1 * inv;
  sl[t] = p0; sl[t + 256] = p1;

  float* imp = out + IMP_OFF + (size_t)b * Nn;
  imp[t] = p0; imp[t + 256] = p1;
  float* msk = out + MASK_OFF + (size_t)b * Nn;
  msk[t] = 0.0f; msk[t + 256] = 0.0f;
  __syncthreads();

  // top-8 by repeated argmax (tie -> lowest index, matching top_k stability)
  for (int k = 0; k < 8; ++k) {
    float a0 = sl[t], a1 = sl[t + 256];
    float bv; int bi;
    if (a0 >= a1) { bv = a0; bi = t; } else { bv = a1; bi = t + 256; }
    sv[t] = bv; si[t] = bi; __syncthreads();
    for (int s = 128; s > 0; s >>= 1) {
      if (t < s) {
        if (sv[t + s] > sv[t] || (sv[t + s] == sv[t] && si[t + s] < si[t])) {
          sv[t] = sv[t + s]; si[t] = si[t + s];
        }
      }
      __syncthreads();
    }
    if (t == 0) { sidx[k] = si[0]; sl[si[0]] = -1.0f; }
    __syncthreads();
  }

  if (t < 8) {
    int idx = sidx[t];
    msk[idx] = 1.0f;
    int* iout = (int*)(out + IDX_OFF);
    iout[b * 8 + t] = idx;
  }

  // gather selected rows (K x D per batch)
  float* sel = out + SEL_OFF + (size_t)b * Kk * Dd;
  for (int q = t; q < Kk * Dd; q += 256) {
    int k = q >> 9;
    int d = q & 511;
    sel[q] = coeff[((size_t)b * Nn + sidx[k]) * Dd + d];
  }
}

// ---------------------------------------------------------------------------
extern "C" void kernel_launch(void* const* d_in, const int* in_sizes, int n_in,
                              void* d_out, int out_size, void* d_ws, size_t ws_size,
                              hipStream_t stream) {
  const float* coeff = (const float*)d_in[0];   // (B, N, D) f32
  const float* W1    = (const float*)d_in[1];   // (D, H)    f32
  const float* b1    = (const float*)d_in[2];   // (H,)      f32
  const float* W2    = (const float*)d_in[3];   // (H,)      f32
  const float* b2    = (const float*)d_in[4];   // scalar    f32
  const float* base  = (const float*)d_in[5];   // (N,)      f32
  float* out = (float*)d_out;

  __bf16* w1f   = (__bf16*)d_ws;                       // 256 KiB: packed W1 fragments
  float*  logit = (float*)((char*)d_ws + 262144);      // 256 KiB: (B*N) logits

  sfs_pack_w1<<<512, 256, 0, stream>>>(W1, w1f);
  sfs_scorer<<<512, 128, 131072, stream>>>(coeff, w1f, b1, W2, b2, base, logit);
  sfs_softmax_topk<<<128, 256, 0, stream>>>(coeff, logit, out);
}
